// GNNFeatureExtractor_57277683859460
// MI455X (gfx1250) — matrix operand order repaired and checked by reading, not compiled, not run on hardware
//
#include <hip/hip_runtime.h>
#include <hip/hip_bf16.h>

#define B_ 8
#define A_ 64
#define S_ 20
#define N_ 5000
#define E_ 20000
#define D_ 64
#define H_ 4
#define L_ 4
#define EPS_ 1e-5f
#define LEAKY_ 0.2f
#define M_ (B_*N_)     /* 40000 rows */
#define EN_ (E_+N_)    /* 25000 edges incl self-loops */

typedef __attribute__((ext_vector_type(16))) _Float16 v16h;
typedef __attribute__((ext_vector_type(8)))  _Float16 v8h;
typedef __attribute__((ext_vector_type(8)))  float    v8f;

union Frag16 { v16h v; v8h h[2]; };

// ---------------------------------------------------------------- utilities
__device__ __forceinline__ unsigned enc_f32(float f) {
  unsigned u = __float_as_uint(f);
  return (u & 0x80000000u) ? ~u : (u | 0x80000000u);   // order-preserving map
}
__device__ __forceinline__ float dec_f32(unsigned e) {
  unsigned u = (e & 0x80000000u) ? (e & 0x7FFFFFFFu) : ~e;
  return __uint_as_float(u);
}

// Pre-swizzle a K x Nc f32 weight matrix into per-lane-contiguous WMMA B-fragment
// order (f16), zero-padding K up to a multiple of 32.
// Bsw[(((chunk*ntiles)+nt)*32 + lane)*16 + p] = W[(chunk*32 + (lane>>4)*16 + p)*Nc + nt*16 + (lane&15)]
__global__ void k_wswz(const float* __restrict__ W, _Float16* __restrict__ Bsw,
                       int K, int Nc, int total) {
  int i = blockIdx.x * blockDim.x + threadIdx.x;
  if (i >= total) return;
  int p = i & 15;
  int lane = (i >> 4) & 31;
  int rest = i >> 9;
  int ntiles = Nc >> 4;
  int nt = rest % ntiles, chunk = rest / ntiles;
  int k = chunk * 32 + ((lane >> 4) << 4) + p;
  int n = (nt << 4) + (lane & 15);
  Bsw[i] = (k < K) ? (_Float16)W[(size_t)k * Nc + n] : (_Float16)0.f;
}

// ---------------------------------------------------------------- goal MLP
__global__ void k_goal(const float* __restrict__ agvs,
                       const float* __restrict__ Wg1, const float* __restrict__ bg1,
                       const float* __restrict__ gw1, const float* __restrict__ gb1,
                       const float* __restrict__ Wg2, const float* __restrict__ bg2,
                       const float* __restrict__ gw2, const float* __restrict__ gb2,
                       float* __restrict__ goal_out) {
  __shared__ float h1[B_][128];
  __shared__ float h2[B_][64];
  __shared__ float ra[128], rb[128];
  const int t = threadIdx.x;           // 128 threads
  for (int b = 0; b < B_; ++b) {
    const float* g = agvs + (size_t)b * A_ * 18 + 4;
    float s = bg1[t];
    for (int j = 0; j < 4; ++j) s += g[j] * Wg1[j * 128 + t];
    h1[b][t] = fmaxf(s, 0.f);
  }
  __syncthreads();
  float sum = 0.f, sq = 0.f;
  for (int b = 0; b < B_; ++b) { float v = h1[b][t]; sum += v; sq += v * v; }
  ra[t] = sum; rb[t] = sq; __syncthreads();
  for (int st = 64; st > 0; st >>= 1) {
    if (t < st) { ra[t] += ra[t + st]; rb[t] += rb[t + st]; }
    __syncthreads();
  }
  {
    const float cnt = (float)(B_ * 128);
    float mean = ra[0] / cnt;
    float var  = rb[0] / cnt - mean * mean;
    float sc   = gw1[0] * rsqrtf(var + EPS_);
    float sh   = gb1[0] - mean * sc;
    for (int b = 0; b < B_; ++b) h1[b][t] = h1[b][t] * sc + sh;
  }
  __syncthreads();
  if (t < 64) {
    for (int b = 0; b < B_; ++b) {
      float s = bg2[t];
      for (int j = 0; j < 128; ++j) s += h1[b][j] * Wg2[j * 64 + t];
      h2[b][t] = fmaxf(s, 0.f);
    }
  }
  __syncthreads();
  sum = 0.f; sq = 0.f;
  if (t < 64) for (int b = 0; b < B_; ++b) { float v = h2[b][t]; sum += v; sq += v * v; }
  ra[t] = sum; rb[t] = sq; __syncthreads();
  for (int st = 64; st > 0; st >>= 1) {
    if (t < st) { ra[t] += ra[t + st]; rb[t] += rb[t + st]; }
    __syncthreads();
  }
  if (t < 64) {
    const float cnt = (float)(B_ * 64);
    float mean = ra[0] / cnt;
    float var  = rb[0] / cnt - mean * mean;
    float sc   = gw2[0] * rsqrtf(var + EPS_);
    float sh   = gb2[0] - mean * sc;
    for (int b = 0; b < B_; ++b) goal_out[b * 64 + t] = h2[b][t] * sc + sh;
  }
}

// ---------------------------------------------------------------- nearest-node argmin
__global__ void k_nidx(const float* __restrict__ agvs, const float* __restrict__ stat,
                       const float* __restrict__ nodes,
                       int* __restrict__ idx_agv, int* __restrict__ idx_stat,
                       int* __restrict__ idx_reach) {
  const int QA = B_ * A_ * 7, QS = B_ * S_, QR = B_ * 4;
  int q = blockIdx.x * blockDim.x + threadIdx.x;
  if (q >= QA + QS + QR) return;
  float cx, cy; int* outp;
  if (q < QA) {
    int b = q / (A_ * 7), r = q % (A_ * 7), a = r / 7, c = r % 7;
    const float* p = agvs + ((size_t)(b * A_ + a)) * 18;
    cx = p[2 + 2 * c]; cy = p[3 + 2 * c]; outp = idx_agv + q;
  } else if (q < QA + QS) {
    int t = q - QA, b = t / S_, s = t % S_;
    cx = stat[(b * S_ + s) * 2]; cy = stat[(b * S_ + s) * 2 + 1]; outp = idx_stat + t;
  } else {
    int t = q - QA - QS, b = t / 4, j = t % 4;
    const float* p = agvs + (size_t)b * A_ * 18;
    cx = p[8 + 2 * j]; cy = p[9 + 2 * j]; outp = idx_reach + t;
  }
  float best = 3.4e38f; int bi = 0;
  for (int n = 0; n < N_; ++n) {
    float dx = nodes[2 * n] - cx, dy = nodes[2 * n + 1] - cy;
    float d = dx * dx + dy * dy;
    if (d < best) { best = d; bi = n; }      // strict < : first-min like argmin
  }
  *outp = bi;
}

// ---------------------------------------------------------------- node features
__global__ void k_feats_init(const float* __restrict__ agvs, const float* __restrict__ nodes,
                             float* __restrict__ featsf) {
  int i = blockIdx.x * blockDim.x + threadIdx.x;
  if (i >= M_) return;
  int b = i / N_, n = i % N_;
  float mx = agvs[(size_t)b * A_ * 18 + 6], my = agvs[(size_t)b * A_ * 18 + 7];
  float dx = nodes[2 * n] - mx, dy = nodes[2 * n + 1] - my;
  float* f = featsf + (size_t)i * 16;
  for (int c = 0; c < 15; ++c) f[c] = 0.f;
  f[15] = fminf(sqrtf(dx * dx + dy * dy), 1.f);
}

__global__ void k_scatter(const float* __restrict__ agvs, const int* __restrict__ idx_agv,
                          const int* __restrict__ idx_stat, float* __restrict__ featsf) {
  const int QA = B_ * A_ * 7, QS = B_ * S_;
  int q = blockIdx.x * blockDim.x + threadIdx.x;
  if (q >= QA + QS) return;
  if (q < QA) {
    int b = q / (A_ * 7), r = q % (A_ * 7), a = r / 7, c = r % 7;
    float dp = agvs[((size_t)(b * A_ + a)) * 18 + 16];
    float val = (c == 0) ? (1.f - dp) : ((c == 1) ? dp : 1.f);
    int col = (a == 0) ? c : (7 + c);
    int node = idx_agv[q];
    atomicAdd(&featsf[((size_t)b * N_ + node) * 16 + col], val);
  } else {
    int t = q - QA, b = t / S_;
    int node = idx_stat[t];
    atomicAdd(&featsf[((size_t)b * N_ + node) * 16 + 14], 1.f);
  }
}

// clip to [0,1], convert to f16, pad row stride 16 -> 32 with zeros
__global__ void k_clip16(const float* __restrict__ src, _Float16* __restrict__ dst) {
  int i = blockIdx.x * blockDim.x + threadIdx.x;
  if (i >= M_ * 32) return;
  int row = i >> 5, c = i & 31;
  dst[i] = (c < 16) ? (_Float16)fminf(fmaxf(src[(size_t)row * 16 + c], 0.f), 1.f)
                    : (_Float16)0.f;
}

// ---------------------------------------------------------------- WMMA GEMM
// C(M x Nc) = act(A_f16(M x Kp) @ Bsw + bias). Kp compile-time, multiple of 32,
// A row-major with stride Kp (zero padded); Bsw pre-swizzled fragment order.
// One wave owns one 16x16 C tile; EXEC is all-ones at every WMMA.
template<int Kp, int Nc>
__global__ void k_gemm(const _Float16* __restrict__ A, const _Float16* __restrict__ Bsw,
                       const float* __restrict__ bias, float* __restrict__ C,
                       int M, int relu) {
  const int lane = threadIdx.x & 31;
  const int wave = threadIdx.x >> 5;
  const int tile = blockIdx.x * (blockDim.x >> 5) + wave;
  constexpr int ntiles = Nc >> 4;
  constexpr int chunks = Kp >> 5;
  const int total = (M >> 4) * ntiles;
  if (tile >= total) return;
  const int mt = tile / ntiles, nt = tile % ntiles;
  const int m  = (mt << 4) + (lane & 15);
  const int hi = lane >> 4;
  const _Float16* arow = A + (size_t)m * Kp + (hi << 4);
  const _Float16* brow = Bsw + ((size_t)nt * 32 + lane) * 16;
  v8f acc = {};
#pragma unroll
  for (int c = 0; c < chunks; ++c) {
    Frag16 fa, fb;
    // A fragment: 16 contiguous halfs (k = 16*hi + s) as two global_load_b128
    fa.h[0] = *(const v8h*)(arow + c * 32);
    fa.h[1] = *(const v8h*)(arow + c * 32 + 8);
    // B fragment: 16 contiguous halfs (32B) in swizzled buffer
    fb.v = *(const v16h*)(brow + (size_t)c * ntiles * 512);
    acc = __builtin_amdgcn_wmma_f32_16x16x32_f16(false, fa.v, false, fb.v,
                                                 (short)0, acc, false, false);
  }
  const int n = (nt << 4) + (lane & 15);
  float bv = bias ? bias[n] : 0.f;
#pragma unroll
  for (int j = 0; j < 8; ++j) {              // C: VGPR j -> row j + hi*8
    int row = (mt << 4) + (hi << 3) + j;
    float v = acc[j] + bv;
    if (relu) v = fmaxf(v, 0.f);
    C[(size_t)row * Nc + n] = v;
  }
}

// ---------------------------------------------------------------- per-node BatchNorm (mean over B x F)
__global__ void k_bn(const float* __restrict__ y, const float* __restrict__ w,
                     const float* __restrict__ bvec, _Float16* __restrict__ out16, int F) {
  __shared__ float rs[256], rq[256];
  const int n = blockIdx.x, t = threadIdx.x;
  const int cnt = B_ * F;
  float s = 0.f, q = 0.f;
  for (int i = t; i < cnt; i += 256) {
    int b = i / F, f = i % F;
    float v = y[((size_t)b * N_ + n) * F + f];
    s += v; q += v * v;
  }
  rs[t] = s; rq[t] = q; __syncthreads();
  for (int st = 128; st > 0; st >>= 1) {
    if (t < st) { rs[t] += rs[t + st]; rq[t] += rq[t + st]; }
    __syncthreads();
  }
  float mean = rs[0] / cnt;
  float var  = rq[0] / cnt - mean * mean;
  float sc   = w[n] * rsqrtf(var + EPS_);
  float sh   = bvec[n] - mean * sc;
  for (int i = t; i < cnt; i += 256) {
    int b = i / F, f = i % F;
    float v = y[((size_t)b * N_ + n) * F + f];
    out16[((size_t)b * N_ + n) * F + f] = (_Float16)(v * sc + sh);
  }
}

// ---------------------------------------------------------------- GAT attention scores + init
__global__ void k_esed(const float* __restrict__ hbuf, const float* __restrict__ As,
                       const float* __restrict__ Ad, float* __restrict__ es,
                       float* __restrict__ ed, unsigned* __restrict__ mmax,
                       float* __restrict__ denom, float* __restrict__ outacc) {
  int i = blockIdx.x * blockDim.x + threadIdx.x;
  if (i >= M_ * H_) return;
  int h = i & 3, bn = i >> 2;
  const float* hp = hbuf + (size_t)bn * 256 + h * 64;
  float s0 = 0.f, s1 = 0.f;
  for (int d = 0; d < 64; ++d) { float hv = hp[d]; s0 += hv * As[h * 64 + d]; s1 += hv * Ad[h * 64 + d]; }
  es[i] = s0; ed[i] = s1;
  mmax[i] = 0u;                 // encoded -inf
  denom[i] = 0.f;
  float* oa = outacc + (size_t)bn * 64 + h * 16;
  for (int d = 0; d < 16; ++d) oa[d] = 0.f;
}

__device__ __forceinline__ void edge_sd(const int* eidx, int e, int& s, int& d) {
  if (e < E_) { s = eidx[e]; d = eidx[E_ + e]; } else { s = d = e - E_; }
}

__global__ void k_edge_max(const float* __restrict__ es, const float* __restrict__ ed,
                           const int* __restrict__ eidx, unsigned* __restrict__ mmax) {
  int i = blockIdx.x * blockDim.x + threadIdx.x;
  if (i >= B_ * EN_) return;
  int b = i / EN_, e = i % EN_, s, d;
  edge_sd(eidx, e, s, d);
  size_t sb = ((size_t)b * N_ + s) * H_, db = ((size_t)b * N_ + d) * H_;
  for (int h = 0; h < H_; ++h) {
    float v = es[sb + h] + ed[db + h];
    v = (v > 0.f) ? v : LEAKY_ * v;
    atomicMax(&mmax[db + h], enc_f32(v));
  }
}

__global__ void k_edge_exp(const float* __restrict__ es, const float* __restrict__ ed,
                           const int* __restrict__ eidx, const unsigned* __restrict__ mmax,
                           float* __restrict__ exbuf, float* __restrict__ denom) {
  int i = blockIdx.x * blockDim.x + threadIdx.x;
  if (i >= B_ * EN_) return;
  int b = i / EN_, e = i % EN_, s, d;
  edge_sd(eidx, e, s, d);
  size_t sb = ((size_t)b * N_ + s) * H_, db = ((size_t)b * N_ + d) * H_;
  for (int h = 0; h < H_; ++h) {
    float v = es[sb + h] + ed[db + h];
    v = (v > 0.f) ? v : LEAKY_ * v;
    float ex = __expf(v - dec_f32(mmax[db + h]));
    exbuf[(size_t)i * H_ + h] = ex;
    atomicAdd(&denom[db + h], ex);
  }
}

__global__ void k_edge_agg(const float* __restrict__ exbuf, const float* __restrict__ denom,
                           const int* __restrict__ eidx, const float* __restrict__ hbuf,
                           float* __restrict__ outacc) {
  int i = blockIdx.x * blockDim.x + threadIdx.x;
  if (i >= B_ * EN_ * H_) return;
  int h = i & 3, be = i >> 2;
  int b = be / EN_, e = be % EN_, s, d;
  edge_sd(eidx, e, s, d);
  float alpha = exbuf[(size_t)be * H_ + h] / denom[((size_t)b * N_ + d) * H_ + h];
  const float* hp = hbuf + ((size_t)b * N_ + s) * 256 + h * 64;
  float* op = outacc + ((size_t)b * N_ + d) * 64;
  for (int dd = 0; dd < 64; ++dd) atomicAdd(&op[dd], alpha * hp[dd]);
}

// ---------------------------------------------------------------- head-mean + bias + GraphNorm
__global__ void k_gat_fin(const float* __restrict__ outacc, const float* __restrict__ gb,
                          const float* __restrict__ w, const float* __restrict__ bb,
                          const float* __restrict__ ms, float* __restrict__ x,
                          _Float16* __restrict__ xg16) {
  int row = blockIdx.x * (blockDim.x >> 5) + (threadIdx.x >> 5);
  int lane = threadIdx.x & 31;
  if (row >= M_) return;
  const float* op = outacc + (size_t)row * 64;
  float t0 = op[lane] * 0.25f + gb[lane];
  float t1 = op[lane + 32] * 0.25f + gb[lane + 32];
  float s = t0 + t1;
  for (int m = 16; m > 0; m >>= 1) s += __shfl_xor(s, m, 32);
  float mean = s * (1.f / 64.f);
  float sh0 = t0 - ms[lane] * mean;
  float sh1 = t1 - ms[lane + 32] * mean;
  float q = sh0 * sh0 + sh1 * sh1;
  for (int m = 16; m > 0; m >>= 1) q += __shfl_xor(q, m, 32);
  float inv = rsqrtf(q * (1.f / 64.f) + EPS_);
  float o0 = w[lane] * sh0 * inv + bb[lane];
  float o1 = w[lane + 32] * sh1 * inv + bb[lane + 32];
  x[(size_t)row * 64 + lane] = o0;
  x[(size_t)row * 64 + lane + 32] = o1;
  xg16[(size_t)row * 64 + lane] = (_Float16)o0;
  xg16[(size_t)row * 64 + lane + 32] = (_Float16)o1;
}

// ---------------------------------------------------------------- final gather
__global__ void k_out(const float* __restrict__ x, const int* __restrict__ idx_reach,
                      const float* __restrict__ goal_out, const float* __restrict__ agvs,
                      float* __restrict__ out) {
  int o = blockIdx.x * blockDim.x + threadIdx.x;
  if (o >= B_ * 321) return;
  int b = o / 321, r = o % 321;
  float v;
  if (r < 256) {
    int j = r >> 6, dd = r & 63;
    int node = idx_reach[b * 4 + j];
    v = (node == N_ - 1) ? 0.f : x[((size_t)b * N_ + node) * 64 + dd];  // x[:,-1]=0
  } else if (r < 320) {
    v = goal_out[b * 64 + (r - 256)];
  } else {
    v = agvs[(size_t)b * A_ * 18 + 17];      // invalid_state
  }
  out[o] = v;
}

static inline int cdiv(int a, int b) { return (a + b - 1) / b; }

extern "C" void kernel_launch(void* const* d_in, const int* in_sizes, int n_in,
                              void* d_out, int out_size, void* d_ws, size_t ws_size,
                              hipStream_t stream) {
  (void)in_sizes; (void)n_in; (void)out_size; (void)ws_size;
  const float* agvs  = (const float*)d_in[0];
  const float* stat  = (const float*)d_in[1];
  const float* nodes = (const float*)d_in[2];
  const int*   eidx  = (const int*)d_in[3];
  const float* Wg1 = (const float*)d_in[4];
  const float* bg1 = (const float*)d_in[5];
  const float* gbnw1 = (const float*)d_in[6];
  const float* gbnb1 = (const float*)d_in[7];
  const float* Wg2 = (const float*)d_in[8];
  const float* bg2 = (const float*)d_in[9];
  const float* gbnw2 = (const float*)d_in[10];
  const float* gbnb2 = (const float*)d_in[11];
  const float* Wn1 = (const float*)d_in[12];
  const float* bn1 = (const float*)d_in[13];
  const float* nbnw1 = (const float*)d_in[14];
  const float* nbnb1 = (const float*)d_in[15];
  const float* Wn2 = (const float*)d_in[16];
  const float* bn2 = (const float*)d_in[17];
  const float* nbnw2 = (const float*)d_in[18];
  const float* nbnb2 = (const float*)d_in[19];
  const float* gatW  = (const float*)d_in[20];
  const float* gatAs = (const float*)d_in[21];
  const float* gatAd = (const float*)d_in[22];
  const float* gatB  = (const float*)d_in[23];
  const float* gnw   = (const float*)d_in[24];
  const float* gnb   = (const float*)d_in[25];
  const float* gnms  = (const float*)d_in[26];
  float* out = (float*)d_out;

  char* base = (char*)d_ws;
  size_t off = 0;
  auto alloc = [&](size_t bytes) -> char* {
    char* p = base + off;
    off = (off + bytes + 255) & ~(size_t)255;
    return p;
  };
  float*     goal_out  = (float*)alloc((size_t)B_ * 64 * sizeof(float));
  int*       idx_agv   = (int*)alloc((size_t)B_ * A_ * 7 * sizeof(int));
  int*       idx_stat  = (int*)alloc((size_t)B_ * S_ * sizeof(int));
  int*       idx_reach = (int*)alloc((size_t)B_ * 4 * sizeof(int));
  _Float16*  wsw_n1    = (_Float16*)alloc((size_t)32 * 128 * sizeof(_Float16));   // padded K
  _Float16*  wsw_n2    = (_Float16*)alloc((size_t)128 * 64 * sizeof(_Float16));
  _Float16*  wsw_gat   = (_Float16*)alloc((size_t)L_ * 64 * 256 * sizeof(_Float16));
  float*     featsf    = (float*)alloc((size_t)M_ * 16 * sizeof(float));
  _Float16*  feats16p  = (_Float16*)alloc((size_t)M_ * 32 * sizeof(_Float16));    // stride 32, padded
  _Float16*  xh16      = (_Float16*)alloc((size_t)M_ * 128 * sizeof(_Float16));
  _Float16*  xg16      = (_Float16*)alloc((size_t)M_ * 64 * sizeof(_Float16));
  float*     xf        = (float*)alloc((size_t)M_ * 64 * sizeof(float));
  float*     esb       = (float*)alloc((size_t)M_ * H_ * sizeof(float));
  float*     edb       = (float*)alloc((size_t)M_ * H_ * sizeof(float));
  unsigned*  mmax      = (unsigned*)alloc((size_t)M_ * H_ * sizeof(unsigned));
  float*     denom     = (float*)alloc((size_t)M_ * H_ * sizeof(float));
  float*     exbuf     = (float*)alloc((size_t)B_ * EN_ * H_ * sizeof(float));
  float*     big1      = (float*)alloc((size_t)M_ * 256 * sizeof(float)); // y1 then hbuf
  float*     big2      = (float*)alloc((size_t)M_ * 64 * sizeof(float));  // y2 then outacc

  const int T = 256;

  // weight pre-swizzle into WMMA B-fragment order (one-time, tiny)
  k_wswz<<<cdiv(32 * 128, T), T, 0, stream>>>(Wn1, wsw_n1, 16, 128, 32 * 128);
  k_wswz<<<cdiv(128 * 64, T), T, 0, stream>>>(Wn2, wsw_n2, 128, 64, 128 * 64);
  for (int l = 0; l < L_; ++l)
    k_wswz<<<cdiv(64 * 256, T), T, 0, stream>>>(gatW + (size_t)l * 64 * 256,
                                                wsw_gat + (size_t)l * 64 * 256,
                                                64, 256, 64 * 256);

  k_goal<<<1, 128, 0, stream>>>(agvs, Wg1, bg1, gbnw1, gbnb1, Wg2, bg2, gbnw2, gbnb2, goal_out);
  k_nidx<<<cdiv(B_ * A_ * 7 + B_ * S_ + B_ * 4, T), T, 0, stream>>>(agvs, stat, nodes,
                                                                    idx_agv, idx_stat, idx_reach);
  k_feats_init<<<cdiv(M_, T), T, 0, stream>>>(agvs, nodes, featsf);
  k_scatter<<<cdiv(B_ * A_ * 7 + B_ * S_, T), T, 0, stream>>>(agvs, idx_agv, idx_stat, featsf);
  k_clip16<<<cdiv(M_ * 32, T), T, 0, stream>>>(featsf, feats16p);

  // node MLP (WMMA GEMMs + per-node BN)
  float* y1 = big1;
  k_gemm<32, 128><<<cdiv((M_ / 16) * (128 / 16), 8), T, 0, stream>>>(feats16p, wsw_n1, bn1, y1, M_, 1);
  k_bn<<<N_, T, 0, stream>>>(y1, nbnw1, nbnb1, xh16, 128);
  float* y2 = big2;
  k_gemm<128, 64><<<cdiv((M_ / 16) * (64 / 16), 8), T, 0, stream>>>(xh16, wsw_n2, bn2, y2, M_, 1);
  k_bn<<<N_, T, 0, stream>>>(y2, nbnw2, nbnb2, xg16, 64);

  // GAT layers
  float* hbuf = big1;
  float* outacc = big2;
  for (int l = 0; l < L_; ++l) {
    k_gemm<64, 256><<<cdiv((M_ / 16) * (256 / 16), 8), T, 0, stream>>>(
        xg16, wsw_gat + (size_t)l * 64 * 256, nullptr, hbuf, M_, 0);
    k_esed<<<cdiv(M_ * H_, T), T, 0, stream>>>(hbuf, gatAs + l * H_ * D_, gatAd + l * H_ * D_,
                                               esb, edb, mmax, denom, outacc);
    k_edge_max<<<cdiv(B_ * EN_, T), T, 0, stream>>>(esb, edb, eidx, mmax);
    k_edge_exp<<<cdiv(B_ * EN_, T), T, 0, stream>>>(esb, edb, eidx, mmax, exbuf, denom);
    k_edge_agg<<<cdiv(B_ * EN_ * H_, T), T, 0, stream>>>(exbuf, denom, eidx, hbuf, outacc);
    k_gat_fin<<<cdiv(M_, 8), T, 0, stream>>>(outacc, gatB + l * D_, gnw + l * D_,
                                             gnb + l * D_, gnms + l * D_, xf, xg16);
  }

  k_out<<<cdiv(B_ * 321, T), T, 0, stream>>>(xf, idx_reach, goal_out, agvs, out);
}
